// SDT_46488726012141
// MI455X (gfx1250) — matrix operand order repaired and checked by reading, not compiled, not run on hardware
//
#include <hip/hip_runtime.h>
#include <hip/hip_bf16.h>
#include <math.h>

// ---- problem constants (from reference) ----
#define DEPTH      10
#define INPUT_DIM  128
#define OUTPUT_DIM 32
#define BATCH      65536
#define INNER      1023          // 2^10 - 1
#define LEAF       1024
#define NPAD       1024          // INNER padded to 1024 columns
#define LAMDA      0.001f

typedef _Float16 v16h  __attribute__((ext_vector_type(16)));
typedef __bf16   v16bf __attribute__((ext_vector_type(16)));
typedef float    v8f   __attribute__((ext_vector_type(8)));
typedef float    v4f   __attribute__((ext_vector_type(4)));

#define PSTRIDE 1028   // 1024 + 4 pad: row stride shifts banks by 4 -> conflict-free column walks

// fast sigmoid: v_exp_f32 + v_add_f32 + v_rcp_f32 (no IEEE div chain)
__device__ __forceinline__ float fast_sigmoid(float x) {
    return __builtin_amdgcn_rcpf(1.0f + __expf(-x));
}

// ============================================================================
// Kernel 0: prep — convert W_inner -> f16 (padded), W_leaf -> bf16, zero accs
// ============================================================================
__global__ __launch_bounds__(256) void sdt_prep(
    const float* __restrict__ Wi, const float* __restrict__ Wl,
    _Float16* __restrict__ W16, __bf16* __restrict__ WLb,
    float* __restrict__ accL, float* __restrict__ accR)
{
    int idx = blockIdx.x * blockDim.x + threadIdx.x;
    if (idx < NPAD * INPUT_DIM) {
        int row = idx >> 7;                       // /128
        float v = (row < INNER) ? Wi[idx] : 0.0f; // pad row 1023 with zeros
        W16[idx] = (_Float16)v;
    }
    if (idx < OUTPUT_DIM * LEAF) {
        WLb[idx] = (__bf16)Wl[idx];
    }
    if (idx < NPAD) {
        accL[idx] = 0.0f;
        accR[idx] = 0.0f;
    }
}

// ============================================================================
// Kernel 1: fused  sigmoid(data @ Wi^T)  ->  tree mu + node sums  ->  mu @ Wl^T
// One workgroup = 16 batch rows, 128 threads = 4 waves (wave32).
// ============================================================================
__global__ __launch_bounds__(128) void sdt_main(
    const float*    __restrict__ data,
    const _Float16* __restrict__ W16,
    const __bf16*   __restrict__ WLb,
    float*          __restrict__ out,
    float*          __restrict__ accL,
    float*          __restrict__ accR)
{
    __shared__ float pbuf [16][PSTRIDE];   // sigmoid outputs, 16 rows x 1024 nodes
    __shared__ float mubuf[16][PSTRIDE];   // path probabilities -> leaf mu

    const int tid  = threadIdx.x;
    const int wave = tid >> 5;
    const int lane = tid & 31;
    const int lrow = lane & 15;            // N (or M) index within 16
    const int lhi  = lane >> 4;            // 0: lanes 0-15, 1: lanes 16-31
    const int m0   = blockIdx.x * 16;      // batch row tile

    // -------- Phase 1: GEMM1 (f16 WMMA) + sigmoid --------
    // A fragment (16x32 f16): lane holds row (m0+lrow),
// K = base..base+15 (contiguous), base = lhi*16, per 32-wide K tile.
    v16h afrag[4];
    {
        const float* drow = data + (size_t)(m0 + lrow) * INPUT_DIM;
        const int kbase = lhi * 16;
        #pragma unroll
        for (int kt = 0; kt < 4; ++kt) {
            const int k0 = kt * 32 + kbase;
            v4f x0 = *(const v4f*)(drow + k0);
            v4f x1 = *(const v4f*)(drow + k0 + 4);
            v4f x2 = *(const v4f*)(drow + k0 + 8);
            v4f x3 = *(const v4f*)(drow + k0 + 12);
            v16h a;
            #pragma unroll
            for (int j = 0; j < 4; ++j) {
                a[j]      = (_Float16)x0[j];
                a[4 + j]  = (_Float16)x1[j];
                a[8 + j]  = (_Float16)x2[j];
                a[12 + j] = (_Float16)x3[j];
            }
            afrag[kt] = a;
        }
    }

    // Each wave covers 16 N-tiles (256 inner-node columns); 4 waves -> 1024.
    for (int t = 0; t < 16; ++t) {
        const int n0 = (wave * 16 + t) * 16;
        // B fragment (32x16 f16): lane holds col (n0+lrow), K = kt*32 + lhi*16 + 0..15
        const _Float16* wrow = W16 + (size_t)(n0 + lrow) * INPUT_DIM + lhi * 16;
        v8f acc = {};
        #pragma unroll
        for (int kt = 0; kt < 4; ++kt) {
            v16h b = *(const v16h*)(wrow + kt * 32);
            acc = __builtin_amdgcn_wmma_f32_16x16x32_f16(
                false, afrag[kt], false, b, (short)0, acc, false, false);
        }
        // C layout: VGPR r, lane -> M = r + lhi*8, N = lrow
        #pragma unroll
        for (int r = 0; r < 8; ++r) {
            pbuf[r + lhi * 8][n0 + lrow] = fast_sigmoid(acc[r]);
        }
    }
    __syncthreads();

    // -------- Phase 2: tree expansion + per-node batch sums --------
    // Level d parents live at spread index j << (10-d); children written in place:
    // left child overwrites parent slot, right child at +2^(9-d).
    for (int d = 0; d < DEPTH; ++d) {
        const int n     = 1 << d;
        const int begin = n - 1;
        const int half  = 1 << (9 - d);
        for (int j = tid; j < n; j += 128) {
            const int idx = j << (10 - d);
            float sL = 0.0f, sR = 0.0f;
            #pragma unroll 4
            for (int r = 0; r < 16; ++r) {
                float mp = (d == 0) ? 1.0f : mubuf[r][idx];
                float pv = pbuf[r][begin + j];
                float l  = mp * pv;
                float rt = mp - l;             // mp * (1 - pv)
                mubuf[r][idx]        = l;
                mubuf[r][idx + half] = rt;
                sL += l; sR += rt;
            }
            atomicAdd(&accL[begin + j], sL);   // sum_B mu*p       (alpha numerator, left)
            atomicAdd(&accR[begin + j], sR);   // sum_B mu*(1-p)   (alpha numerator, right)
        }
        __syncthreads();
    }

    // -------- Phase 3: GEMM2  out = mu @ W_leaf^T  (bf16 WMMA) --------
    // bf16 keeps f32 exponent range (leaf mu can be ~1e-30; f16 would flush).
    // 16x32 output tile: waves 0 and 1 each take one 16-wide N tile, K = 1024.
    if (wave < 2) {
        const int n0 = wave * 16;
        const __bf16* wl = WLb + (size_t)(n0 + lrow) * LEAF + lhi * 16;
        v8f acc = {};
        for (int kt = 0; kt < 32; ++kt) {
            // A fragment from LDS mu (f32 -> bf16): row lrow, K = lhi*16 + 0..15 (contiguous)
            const int kb = kt * 32 + lhi * 16;
            v16bf a;
            #pragma unroll
            for (int j = 0; j < 8; ++j) {
                a[j]     = (__bf16)mubuf[lrow][kb + j];
                a[8 + j] = (__bf16)mubuf[lrow][kb + 8 + j];
            }
            v16bf b = *(const v16bf*)(wl + kt * 32);
            acc = __builtin_amdgcn_wmma_f32_16x16x32_bf16(
                false, a, false, b, (short)0, acc, false, false);
        }
        #pragma unroll
        for (int r = 0; r < 8; ++r) {
            out[(size_t)(m0 + r + lhi * 8) * OUTPUT_DIM + n0 + lrow] = acc[r];
        }
    }
}

// ============================================================================
// Kernel 2: penalty from node statistics.
// den(g) = sum_B mu_g = B for root, else the parent's directional numerator.
// ============================================================================
__global__ __launch_bounds__(256) void sdt_penalty(
    const float* __restrict__ accL, const float* __restrict__ accR,
    float* __restrict__ out_pen)
{
    __shared__ float red[256];
    const int tid = threadIdx.x;
    float s = 0.0f;
    for (int g = tid; g < INNER; g += 256) {
        int d = 31 - __clz(g + 1);             // node depth (layer)
        float den = (g == 0) ? (float)BATCH
                             : ((g & 1) ? accL[(g - 1) >> 1] : accR[(g - 1) >> 1]);
        float inv = 1.0f / den;
        float aL = accL[g] * inv;
        float aR = accR[g] * inv;
        float c = logf(aL) + log1pf(-aL) + logf(aR) + log1pf(-aR);
        s += -LAMDA * exp2f((float)(-d)) * 0.5f * c;
    }
    red[tid] = s;
    __syncthreads();
    for (int off = 128; off > 0; off >>= 1) {
        if (tid < off) red[tid] += red[tid + off];
        __syncthreads();
    }
    if (tid == 0) *out_pen = red[0];
}

// ============================================================================
// Launch
// ============================================================================
extern "C" void kernel_launch(void* const* d_in, const int* in_sizes, int n_in,
                              void* d_out, int out_size, void* d_ws, size_t ws_size,
                              hipStream_t stream)
{
    const float* data    = (const float*)d_in[0];   // (65536, 128)
    const float* W_inner = (const float*)d_in[1];   // (1023, 128)
    const float* W_leaf  = (const float*)d_in[2];   // (32, 1024)
    float* out = (float*)d_out;                     // 65536*32 outputs + 1 penalty

    // workspace layout
    char* ws = (char*)d_ws;
    _Float16* W16  = (_Float16*)(ws);                              // 1024*128*2 = 262144 B
    __bf16*   WLb  = (__bf16*)(ws + 262144);                       //   32*1024*2 =  65536 B
    float*    accL = (float*)(ws + 262144 + 65536);                // 1024*4
    float*    accR = (float*)(ws + 262144 + 65536 + 4096);         // 1024*4

    // prep: convert weights, zero accumulators (must re-init every call)
    sdt_prep<<<(NPAD * INPUT_DIM + 255) / 256, 256, 0, stream>>>(
        W_inner, W_leaf, W16, WLb, accL, accR);

    // fused main pass: 4096 tiles of 16 batch rows
    sdt_main<<<BATCH / 16, 128, 0, stream>>>(data, W16, WLb, out, accL, accR);

    // penalty scalar
    sdt_penalty<<<1, 256, 0, stream>>>(accL, accR, out + (size_t)BATCH * OUTPUT_DIM);
}
